// GraphCategorical_22308060136193
// MI455X (gfx1250) — matrix operand order repaired and checked by reading, not compiled, not run on hardware
//
#include <hip/hip_runtime.h>
#include <hip/hip_bf16.h>

// ---------------------------------------------------------------------------
// GraphCategorical policy net on MI455X (gfx1250), wave32 + bf16 WMMA.
// All heavy GEMMs use v_wmma_f32_16x16x32_bf16 (f32 accumulate), 128x128
// block tiles, B-fragments pre-swizzled in LDS, next-chunk global_prefetch.
// GAT attention is flash-style: logits leaky(src_i+dst_j)+mask_j computed on
// the fly (softmax shift-invariance removes the valid_i term), probabilities
// converted to bf16 A-fragments and multiplied against the per-(b,head)
// 512x64 value slice held in LDS pre-swizzled into B-fragment layout.
// ---------------------------------------------------------------------------

typedef __bf16 bf16;
typedef bf16  v8bf  __attribute__((ext_vector_type(8)));
typedef bf16  v16bf __attribute__((ext_vector_type(16)));
typedef float v8f   __attribute__((ext_vector_type(8)));

#define B_     64
#define N_     512
#define DIN_   512
#define H_     256
#define Z_     128
#define HEADS_ 4
#define HD_    64
#define BN_    (B_ * N_)          // 32768 rows

__device__ __forceinline__ float lrelu(float x) { return x > 0.f ? x : 0.2f * x; }

__device__ __forceinline__ v16bf combine16(v8bf lo, v8bf hi) {
  v16bf r;
#pragma unroll
  for (int i = 0; i < 8; ++i) { r[i] = lo[i]; r[i + 8] = hi[i]; }
  return r;
}

// ---------------------------------------------------------------------------
// maskf[b*N+n] = add_input < 0 ? -1e9 : 0
// ---------------------------------------------------------------------------
__global__ __launch_bounds__(256) void mask_kernel(const int* __restrict__ add,
                                                   float* __restrict__ maskf) {
  int idx = blockIdx.x * 256 + threadIdx.x;
  maskf[idx] = (add[idx] < 0) ? -1e9f : 0.f;
}

// ---------------------------------------------------------------------------
// s = relu(x @ Ws1 + bs1) @ Ws2 + bs2       (tiny: one block per batch row)
// ---------------------------------------------------------------------------
__global__ __launch_bounds__(256) void state_kernel(
    const float* __restrict__ x, const float* __restrict__ Ws1,
    const float* __restrict__ bs1, const float* __restrict__ Ws2,
    const float* __restrict__ bs2, float* __restrict__ s) {
  __shared__ float xs[DIN_];
  __shared__ float hid[H_];
  const int b = blockIdx.x, t = threadIdx.x;
  xs[t] = x[(size_t)b * DIN_ + t];
  xs[t + 256] = x[(size_t)b * DIN_ + t + 256];
  __syncthreads();
  float acc = bs1[t];
  for (int k = 0; k < DIN_; ++k) acc += xs[k] * Ws1[(size_t)k * H_ + t];
  hid[t] = fmaxf(acc, 0.f);
  __syncthreads();
  float acc2 = bs2[t];
  for (int k = 0; k < H_; ++k) acc2 += hid[k] * Ws2[(size_t)k * H_ + t];
  s[(size_t)b * H_ + t] = acc2;
}

// ---------------------------------------------------------------------------
// Generic bf16 WMMA GEMM:  C[BN,256] = A[BN,KDIM] * W[KDIM,256] (+bias)(+relu)
//                          (+resid) -> f32 and/or bf16 outputs.
// AMODE: 0 = bf16 A buffer, 1 = f32 A buffer (convert), 2 = gather emb[ids].
// Block: 256 thr (8 waves), tile 128 rows x 128 cols, K-chunks of 32.
// Each wave owns a 16-row strip across all 128 cols (8 accumulators), so 8
// back-to-back WMMAs per barrier with all B-fragments preloaded first.
// A frag (16x32 bf16): lane l: m=l&15, kb=(l<16)?0:16; halfword h ->
//   k = kb+h  (two contiguous 16B LDS reads).
// B frag (32x16 bf16): lane l: n=l&15, kb=(l<16)?0:16; halfword h -> k=kb+h;
//   W tile is swizzled into that order at LDS-store time.
// ---------------------------------------------------------------------------
template <int KDIM, int AMODE, bool BIAS, bool RELU, bool RESID, int OMODE>
__global__ __launch_bounds__(256) void gemm_kernel(
    const bf16* __restrict__ Abf, const float* __restrict__ Af32,
    const int* __restrict__ ids, const float* __restrict__ W,
    const float* __restrict__ bias, const float* __restrict__ resid,
    float* __restrict__ outF, bf16* __restrict__ outB) {
  __shared__ __align__(16) bf16 As[128][40];     // 32 K + 8 pad (16B-aligned rows)
  __shared__ __align__(16) bf16 Bs[8][32][16];   // frag-order swizzled, 8KB

  const int tid = threadIdx.x;
  const int rowbase = blockIdx.x * 128;
  const int cb = blockIdx.y * 128;

  const int w = tid >> 5, l = tid & 31;
  const int m = l & 15;
  const int kb_a = (l < 16) ? 0 : 16;
  const int arow = w * 16 + m;

  // loader roles
  const int lrow = tid >> 1;               // A: 2 threads per row
  const int kh = (tid & 1) * 16;           // 16 k's each
  const int bkk = tid >> 3;                // B: k row 0..31
  const int bnn = (tid & 7) * 16;          // 16 cols each

  v8f acc[8] = {};

  for (int kc = 0; kc < KDIM / 32; ++kc) {
    // ---- stage A tile (128 x 32 bf16) ----
    {
      const int gr = rowbase + lrow;
      if (AMODE == 0) {
        const bf16* ap = Abf + (size_t)gr * KDIM + kc * 32 + kh;
        *(v8bf*)&As[lrow][kh]     = *(const v8bf*)(ap);
        *(v8bf*)&As[lrow][kh + 8] = *(const v8bf*)(ap + 8);
        if (kc + 1 < KDIM / 32) __builtin_prefetch(ap + 32, 0, 1);
      } else {
        const float* ap;
        if (AMODE == 2) {
          int id = ids[gr];
          if (id < 0) id = 0;                            // aval = where(mask,0,id)
          ap = Af32 + (size_t)id * KDIM + kc * 32 + kh;  // emb gather
        } else {
          ap = Af32 + (size_t)gr * KDIM + kc * 32 + kh;
        }
        if (kc + 1 < KDIM / 32) __builtin_prefetch(ap + 32, 0, 1);
        v8bf t0, t1;
#pragma unroll
        for (int i = 0; i < 8; ++i) { t0[i] = (bf16)ap[i]; t1[i] = (bf16)ap[8 + i]; }
        *(v8bf*)&As[lrow][kh]     = t0;
        *(v8bf*)&As[lrow][kh + 8] = t1;
      }
    }
    // ---- stage B tile (32 x 128 f32 -> bf16, swizzled to frag order) ----
    {
      const float* wp = W + (size_t)(kc * 32 + bkk) * 256 + cb + bnn;
      if (kc + 1 < KDIM / 32) __builtin_prefetch(wp + 32 * 256, 0, 1);
      const int lane_hi = (bkk & 16) ? 16 : 0;
      const int hh = bkk & 15;
#pragma unroll
      for (int i = 0; i < 16; ++i) {
        int n = bnn + i;
        Bs[n >> 4][(n & 15) + lane_hi][hh] = (bf16)wp[i];
      }
    }
    __syncthreads();

    // ---- compute: preload 1 A-frag + 8 B-frags, then 8 WMMAs ----
    v8bf alo = *(const v8bf*)&As[arow][kb_a];
    v8bf ahi = *(const v8bf*)&As[arow][kb_a + 8];
    const v16bf af = combine16(alo, ahi);
    v8bf br[16];
#pragma unroll
    for (int nt = 0; nt < 8; ++nt) {
      br[2 * nt]     = *(const v8bf*)&Bs[nt][l][0];
      br[2 * nt + 1] = *(const v8bf*)&Bs[nt][l][8];
    }
#pragma unroll
    for (int nt = 0; nt < 8; ++nt) {
      const v16bf bfv = combine16(br[2 * nt], br[2 * nt + 1]);
      acc[nt] = __builtin_amdgcn_wmma_f32_16x16x32_bf16(
          false, af, false, bfv, (short)0, acc[nt], false, false);
    }
    __syncthreads();
  }

  // ---- epilogue (C layout: VGPR r -> M=r (lanes 0-15) / M=r+8 (16-31)) ----
  const int n = l & 15;
  const int mofs = (l < 16) ? 0 : 8;
#pragma unroll
  for (int nt = 0; nt < 8; ++nt) {
#pragma unroll
    for (int r = 0; r < 8; ++r) {
      const int row = rowbase + w * 16 + r + mofs;
      const int col = cb + nt * 16 + n;
      float v = acc[nt][r];
      if (BIAS) v += bias[col];
      if (RELU) v = fmaxf(v, 0.f);
      const size_t o = (size_t)row * 256 + col;
      if (RESID) v += resid[o];
      if (OMODE & 1) outF[o] = v;
      if (OMODE & 2) outB[o] = (bf16)v;
    }
  }
}

// ---------------------------------------------------------------------------
// concat(s[b], act[r]) -> LayerNorm(512) -> bf16 ln_out. One wave per row.
// ---------------------------------------------------------------------------
__global__ __launch_bounds__(256) void ln_kernel(
    const float* __restrict__ s, const float* __restrict__ act,
    const float* __restrict__ g, const float* __restrict__ beta,
    bf16* __restrict__ outp) {
  const int w = threadIdx.x >> 5, l = threadIdx.x & 31;
  const size_t r = (size_t)blockIdx.x * 8 + w;
  const int b = (int)(r >> 9);
  const int j0 = l * 16;
  float v[16];
#pragma unroll
  for (int i = 0; i < 16; ++i) {
    int j = j0 + i;
    v[i] = (j < 256) ? s[(size_t)b * 256 + j] : act[r * 256 + (j - 256)];
  }
  float sm = 0.f, sq = 0.f;
#pragma unroll
  for (int i = 0; i < 16; ++i) { sm += v[i]; sq += v[i] * v[i]; }
#pragma unroll
  for (int d = 1; d < 32; d <<= 1) { sm += __shfl_xor(sm, d); sq += __shfl_xor(sq, d); }
  const float mean = sm * (1.f / 512.f);
  const float var = sq * (1.f / 512.f) - mean * mean;
  const float rs = rsqrtf(var + 1e-5f);
#pragma unroll
  for (int i = 0; i < 16; ++i) {
    int j = j0 + i;
    outp[r * 512 + j] = (bf16)((v[i] - mean) * rs * g[j] + beta[j]);
  }
}

// ---------------------------------------------------------------------------
// e_src[b,h,i], e_dst[b,h,i] = <hgat[b,i,h,:], a_src/dst[h,:]>
// ---------------------------------------------------------------------------
__global__ __launch_bounds__(256) void edot_kernel(
    const bf16* __restrict__ hgat, const float* __restrict__ a_src,
    const float* __restrict__ a_dst, float* __restrict__ esrc,
    float* __restrict__ edst) {
  const int flat = blockIdx.x * 256 + threadIdx.x;  // BN * HEADS
  const int h = flat & 3, r = flat >> 2;
  const bf16* hp = hgat + (size_t)r * 256 + h * 64;
  const float* as = a_src + h * 64;
  const float* ad = a_dst + h * 64;
  float s1 = 0.f, s2 = 0.f;
#pragma unroll 8
  for (int d = 0; d < 64; ++d) {
    float x = (float)hp[d];
    s1 += x * as[d];
    s2 += x * ad[d];
  }
  const int b = r >> 9, i = r & 511;
  const size_t o = ((size_t)b * HEADS_ + h) * 512 + i;
  esrc[o] = s1;
  edst[o] = s2;
}

// ---------------------------------------------------------------------------
// Flash GAT: one block per (b, head). Value slice (512x64 bf16 = 64KB) held
// in LDS pre-swizzled into WMMA B-fragment order. Each wave owns 16-row
// i-tiles: two passes compute rowmax/rowsum of leaky(src_i+dst_j)+mask_j,
// then probabilities stream into bf16 A-fragments for v_wmma aggregation.
// All 4 B-fragments of a j-chunk are preloaded before the WMMA chain.
// ---------------------------------------------------------------------------
__global__ __launch_bounds__(256) void attn_kernel(
    const bf16* __restrict__ hgat, const float* __restrict__ esrc,
    const float* __restrict__ edst, const float* __restrict__ maskf,
    bf16* __restrict__ attout) {
  __shared__ __align__(16) bf16 Hs[16][4][32][16];  // [jchunk][nt][lane][h] = 64KB

  const int b = blockIdx.x >> 2;
  const int head = blockIdx.x & 3;
  const int tid = threadIdx.x;

  // stage value slice, swizzled to B-frag order (n=d&15 in lane, k=j%32)
  for (int idx = tid; idx < 512 * 64; idx += 256) {
    const int j = idx >> 6, d = idx & 63;
    const bf16 v = hgat[((size_t)(b * 512 + j)) * 256 + head * 64 + d];
    const int k = j & 31;
    Hs[j >> 5][d >> 4][(d & 15) + ((k & 16) ? 16 : 0)][k & 15] = v;
  }
  __syncthreads();

  const int w = tid >> 5, l = tid & 31;
  const int m = l & 15, half = l >> 4;
  const float* dstp = edst + ((size_t)b * HEADS_ + head) * 512;
  const float* srcp = esrc + ((size_t)b * HEADS_ + head) * 512;
  const float* mp = maskf + (size_t)b * 512;
  const int kb = (l < 16) ? 0 : 16;

  for (int it = w; it < 32; it += 8) {             // 32 i-tiles of 16 rows
    const int i0 = it * 16;
    const float srcv = srcp[i0 + m];               // lane's own row stats
    const int jb = half * 256;

    float mx = -3.4e38f;
    for (int t = 0; t < 256; ++t) {
      const int j = jb + t;
      float x = lrelu(srcv + dstp[j]) + mp[j];
      mx = fmaxf(mx, x);
    }
    mx = fmaxf(mx, __shfl_xor(mx, 16));

    float sme = 0.f;
    for (int t = 0; t < 256; ++t) {
      const int j = jb + t;
      float x = lrelu(srcv + dstp[j]) + mp[j];
      sme += __expf(x - mx);
    }
    sme += __shfl_xor(sme, 16);
    const float inv = 1.f / sme;

    v8f acc[4] = {};
    for (int jc = 0; jc < 16; ++jc) {
      // preload all 4 B-fragments of this j-chunk (8 x ds_load_b128)
      v8bf br[8];
#pragma unroll
      for (int nt = 0; nt < 4; ++nt) {
        br[2 * nt]     = *(const v8bf*)&Hs[jc][nt][l][0];
        br[2 * nt + 1] = *(const v8bf*)&Hs[jc][nt][l][8];
      }
      v16bf af;
#pragma unroll
      for (int hw = 0; hw < 16; ++hw) {
        const int k = kb + hw;
        const int j = jc * 32 + k;
        float x = lrelu(srcv + dstp[j]) + mp[j];
        af[hw] = (bf16)(__expf(x - mx) * inv);
      }
#pragma unroll
      for (int nt = 0; nt < 4; ++nt) {
        const v16bf bfv = combine16(br[2 * nt], br[2 * nt + 1]);
        acc[nt] = __builtin_amdgcn_wmma_f32_16x16x32_bf16(
            false, af, false, bfv, (short)0, acc[nt], false, false);
      }
    }
    const int n = l & 15, mofs = half * 8;
#pragma unroll
    for (int nt = 0; nt < 4; ++nt) {
#pragma unroll
      for (int r = 0; r < 8; ++r) {
        const size_t o =
            ((size_t)(b * 512 + i0 + r + mofs)) * 256 + head * 64 + nt * 16 + n;
        attout[o] = (bf16)acc[nt][r];
      }
    }
  }
}

// ---------------------------------------------------------------------------
// logits[r] = mask ? -1e15 : relu-MLP-out @ Wr2 + br2   (one wave per row)
// ---------------------------------------------------------------------------
__global__ __launch_bounds__(256) void logits_kernel(
    const bf16* __restrict__ rbf, const float* __restrict__ Wr2,
    const float* __restrict__ br2, const int* __restrict__ add,
    float* __restrict__ out) {
  const int w = threadIdx.x >> 5, l = threadIdx.x & 31;
  const size_t r = (size_t)blockIdx.x * 8 + w;
  float p = 0.f;
#pragma unroll
  for (int i = 0; i < 8; ++i) {
    const int k = l * 8 + i;
    p += (float)rbf[r * 256 + k] * Wr2[k];
  }
#pragma unroll
  for (int d = 1; d < 32; d <<= 1) p += __shfl_xor(p, d);
  if (l == 0) out[r] = (add[r] < 0) ? -1e15f : (p + br2[0]);
}

// ---------------------------------------------------------------------------
extern "C" void kernel_launch(void* const* d_in, const int* in_sizes, int n_in,
                              void* d_out, int out_size, void* d_ws,
                              size_t ws_size, hipStream_t stream) {
  (void)in_sizes; (void)n_in; (void)out_size; (void)ws_size;
  const float* x    = (const float*)d_in[0];
  const int*   add  = (const int*)d_in[1];
  const float* emb  = (const float*)d_in[2];
  const float* Wa   = (const float*)d_in[3];
  const float* ba   = (const float*)d_in[4];
  const float* Ws1  = (const float*)d_in[5];
  const float* bs1  = (const float*)d_in[6];
  const float* Ws2  = (const float*)d_in[7];
  const float* bs2  = (const float*)d_in[8];
  const float* ln_g = (const float*)d_in[9];
  const float* ln_b = (const float*)d_in[10];
  const float* Wm1  = (const float*)d_in[11];
  const float* bm1  = (const float*)d_in[12];
  const float* Wm2  = (const float*)d_in[13];
  const float* bm2  = (const float*)d_in[14];
  const float* Wg   = (const float*)d_in[15];
  const float* a_src = (const float*)d_in[16];
  const float* a_dst = (const float*)d_in[17];
  const float* Wo   = (const float*)d_in[18];
  const float* bo   = (const float*)d_in[19];
  const float* Wr1  = (const float*)d_in[20];
  const float* br1  = (const float*)d_in[21];
  const float* Wr2  = (const float*)d_in[22];
  const float* br2  = (const float*)d_in[23];
  float* out = (float*)d_out;

  // workspace carve-up (all offsets 256B-aligned)
  char* ws = (char*)d_ws;
  size_t o = 0;
  float* maskf = (float*)(ws + o); o += (size_t)BN_ * 4;            // 128 KB
  float* sbuf  = (float*)(ws + o); o += (size_t)B_ * H_ * 4;        //  64 KB
  float* bufAf = (float*)(ws + o); o += (size_t)BN_ * 256 * 4;      // act -> nodes f32
  bf16*  lnB   = (bf16*)(ws + o);                                   // ln_out bf16
  bf16*  hgatB = lnB;                                               // reuse after Wm1
  bf16*  attB  = (bf16*)(ws + o + (size_t)BN_ * 256 * 2);
  o += (size_t)BN_ * 512 * 2;
  bf16*  hidB  = (bf16*)(ws + o); o += (size_t)BN_ * 256 * 2;       // also rbf
  bf16*  nodesB = (bf16*)(ws + o); o += (size_t)BN_ * 256 * 2;
  float* esrc = (float*)(ws + o); o += (size_t)B_ * HEADS_ * N_ * 4;
  float* edst = (float*)(ws + o); o += (size_t)B_ * HEADS_ * N_ * 4;

  const dim3 gg(BN_ / 128, 256 / 128);  // GEMM grid: 256 x 2

  mask_kernel<<<BN_ / 256, 256, 0, stream>>>(add, maskf);
  state_kernel<<<B_, 256, 0, stream>>>(x, Ws1, bs1, Ws2, bs2, sbuf);

  // act = emb[aval] @ Wa + ba                (gather-A GEMM, K=128)
  gemm_kernel<128, 2, true, false, false, 1><<<gg, 256, 0, stream>>>(
      nullptr, emb, add, Wa, ba, nullptr, bufAf, nullptr);

  // concat + LayerNorm -> bf16
  ln_kernel<<<BN_ / 8, 256, 0, stream>>>(sbuf, bufAf, ln_g, ln_b, lnB);

  // hidden = relu(ln @ Wm1 + bm1)            (K=512)
  gemm_kernel<512, 0, true, true, false, 2><<<gg, 256, 0, stream>>>(
      lnB, nullptr, nullptr, Wm1, bm1, nullptr, nullptr, hidB);

  // nodes = hidden @ Wm2 + bm2               (f32 + bf16 outputs)
  gemm_kernel<256, 0, true, false, false, 3><<<gg, 256, 0, stream>>>(
      hidB, nullptr, nullptr, Wm2, bm2, nullptr, bufAf, nodesB);

  for (int mp = 0; mp < 2; ++mp) {
    // hgat = nodes @ Wg  (no bias)
    gemm_kernel<256, 0, false, false, false, 2><<<gg, 256, 0, stream>>>(
        nodesB, nullptr, nullptr, Wg, nullptr, nullptr, nullptr, hgatB);
    edot_kernel<<<BN_ * HEADS_ / 256, 256, 0, stream>>>(hgatB, a_src, a_dst,
                                                        esrc, edst);
    attn_kernel<<<B_ * HEADS_, 256, 0, stream>>>(hgatB, esrc, edst, maskf,
                                                 attB);
    // nodes = att @ Wo + bo + nodes  (residual; f32 + bf16 outputs)
    gemm_kernel<256, 0, true, false, true, 3><<<gg, 256, 0, stream>>>(
        attB, nullptr, nullptr, Wo, bo, bufAf, bufAf, nodesB);
  }

  // r = relu(nodes @ Wr1 + br1)
  gemm_kernel<256, 0, true, true, false, 2><<<gg, 256, 0, stream>>>(
      nodesB, nullptr, nullptr, Wr1, br1, nullptr, nullptr, hidB);

  logits_kernel<<<BN_ / 8, 256, 0, stream>>>(hidB, Wr2, br2, add, out);
}
